// AutoregressiveDecoder_39367670235979
// MI455X (gfx1250) — matrix operand order repaired and checked by reading, not compiled, not run on hardware
//
#include <hip/hip_runtime.h>
#include <hip/hip_bf16.h>
#include <math.h>

// ---------------------------------------------------------------------------
// AutoregressiveDecoder forward for MI455X (gfx1250, wave32, WMMA bf16).
// All GEMMs / attention einsums run on v_wmma_f32_16x16x32_bf16.
// Tile staging uses gfx1250 async global->LDS copies (ASYNCcnt) with
// double-buffered LDS software pipelining.
// x (the residual stream) lives in d_out (fp32 [B*T, 512]).
// ---------------------------------------------------------------------------

typedef __bf16 bf16_t;
typedef __attribute__((ext_vector_type(16))) __bf16 bf16x16;
typedef __attribute__((ext_vector_type(8)))  float  f32x8;
typedef __attribute__((ext_vector_type(4)))  int    i32x4v;

#define DIMX 512
#define DEPTHX 2
#define DHX 64
#define CHX 2
#define SHX 8
#define BX 8
#define TX 1024
#define NNX 2048
#define CIX 128
#define SIX 512
#define FFX 2048

// -------------------- gfx1250 async global->LDS (guarded) ------------------
#if defined(__has_builtin)
#if __has_builtin(__builtin_amdgcn_global_load_async_to_lds_b128)
#define HAVE_ASYNC_LDS 1
#endif
#endif
#ifndef HAVE_ASYNC_LDS
#define HAVE_ASYNC_LDS 0
#endif

#if HAVE_ASYNC_LDS
#define AS1 __attribute__((address_space(1)))
#define AS3 __attribute__((address_space(3)))
static __device__ __forceinline__ void async_copy16(const void* g, void* l) {
  // per-lane 16B copy, tracked with ASYNCcnt (ISA ch.8 GLOBAL_LOAD_ASYNC_TO_LDS)
  AS1 i32x4v* gp = (AS1 i32x4v*)(void*)g;
  AS3 i32x4v* lp = (AS3 i32x4v*)l;
  __builtin_amdgcn_global_load_async_to_lds_b128(gp, lp, 0, 0);
}
#if __has_builtin(__builtin_amdgcn_s_wait_asynccnt)
#define ASYNC_WAIT() __builtin_amdgcn_s_wait_asynccnt(0)
#else
#define ASYNC_WAIT() asm volatile("s_wait_asynccnt 0x0" ::: "memory")
#endif
#else
#define ASYNC_WAIT() do { } while (0)
#endif

static __device__ __forceinline__ f32x8 wmma_bf(bf16x16 a, bf16x16 b, f32x8 c) {
  return __builtin_amdgcn_wmma_f32_16x16x32_bf16(false, a, false, b, (short)0, c,
                                                 false, false);
}

// A-matrix fragment (16x32 bf16). ISA 7.12.2: lane<16 row=lane holds K 0..7 &
// 16..23; lane>=16 row=lane-16 holds K 8..15 & 24..31. base is row-major [row][k].
static __device__ __forceinline__ bf16x16 load_a_frag(const bf16_t* base, int ldk,
                                                      int lane) {
  int row = lane & 15;
  int kb  = (lane >> 4) * 8;
  union { uint4 u[2]; bf16x16 v; } t;
  t.u[0] = *(const uint4*)(base + row * ldk + kb);
  t.u[1] = *(const uint4*)(base + row * ldk + kb + 16);
  return t.v;
}

// B-matrix fragment (32x16 bf16). lane<16 col=lane holds K 0..7 & 16..23; lane>=16
// holds K 8..15 & 24..31 (same order as A). base is [n][k].
static __device__ __forceinline__ bf16x16 load_b_frag(const bf16_t* base, int ldk,
                                                      int lane) {
  int col = lane & 15;
  int kb  = (lane >> 4) * 8;
  union { uint4 u[2]; bf16x16 v; } t;
  t.u[0] = *(const uint4*)(base + col * ldk + kb);
  t.u[1] = *(const uint4*)(base + col * ldk + kb + 16);
  return t.v;
}

// ---------------------------------------------------------------------------
// fp32 -> bf16 conversion (weights)
// ---------------------------------------------------------------------------
__global__ void cvt_bf16_kernel(const float* __restrict__ s, bf16_t* __restrict__ d,
                                long n) {
  long i = (long)blockIdx.x * blockDim.x + threadIdx.x;
  long stride = (long)gridDim.x * blockDim.x;
  for (; i < n; i += stride) d[i] = (bf16_t)s[i];
}

// ---------------------------------------------------------------------------
// LayerNorm over 512-wide rows, one wave32 per row, writes bf16
// ---------------------------------------------------------------------------
__global__ __launch_bounds__(256) void ln_bf16_kernel(
    const float* __restrict__ x, const float* __restrict__ w,
    const float* __restrict__ b, bf16_t* __restrict__ y, int rows) {
  int wave = threadIdx.x >> 5, lane = threadIdx.x & 31;
  int row = blockIdx.x * 8 + wave;
  if (row >= rows) return;
  const float* xr = x + (long)row * DIMX;
  float4 v[4];
  float sum = 0.f, ss = 0.f;
#pragma unroll
  for (int j = 0; j < 4; j++) {
    v[j] = *(const float4*)(xr + j * 128 + lane * 4);
    sum += v[j].x + v[j].y + v[j].z + v[j].w;
    ss += v[j].x * v[j].x + v[j].y * v[j].y + v[j].z * v[j].z + v[j].w * v[j].w;
  }
#pragma unroll
  for (int off = 1; off < 32; off <<= 1) {
    sum += __shfl_xor(sum, off, 32);
    ss  += __shfl_xor(ss, off, 32);
  }
  float mean = sum * (1.f / DIMX);
  float var  = ss * (1.f / DIMX) - mean * mean;
  float rstd = rsqrtf(var + 1e-5f);
  bf16_t* yr = y + (long)row * DIMX;
#pragma unroll
  for (int j = 0; j < 4; j++) {
    int c = j * 128 + lane * 4;
    float o0 = (v[j].x - mean) * rstd * w[c + 0] + b[c + 0];
    float o1 = (v[j].y - mean) * rstd * w[c + 1] + b[c + 1];
    float o2 = (v[j].z - mean) * rstd * w[c + 2] + b[c + 2];
    float o3 = (v[j].w - mean) * rstd * w[c + 3] + b[c + 3];
    yr[c + 0] = (bf16_t)o0; yr[c + 1] = (bf16_t)o1;
    yr[c + 2] = (bf16_t)o2; yr[c + 3] = (bf16_t)o3;
  }
}

// ---------------------------------------------------------------------------
// Rotary embedding + head split (+ optional 1/sqrt(DH) pre-scale), fp32 -> bf16
// dst layout [B, H, Tn, 64]
// ---------------------------------------------------------------------------
__global__ void rope_kernel(const float* __restrict__ src, int srcRowLen, int srcOff,
                            const float* __restrict__ freqs, bf16_t* __restrict__ dst,
                            int Hn, int Tn, float scale, int rotate) {
  int idx = blockIdx.x * blockDim.x + threadIdx.x;
  int total = BX * Tn * Hn * 32;
  if (idx >= total) return;
  int p = idx & 31;
  int h = (idx >> 5) % Hn;
  int t = (idx / (32 * Hn)) % Tn;
  int b = idx / (32 * Hn * Tn);
  long srow = (long)b * Tn + t;
  const float* sp = src + srow * srcRowLen + srcOff + h * 64 + 2 * p;
  float x0 = sp[0], x1 = sp[1];
  float y0 = x0, y1 = x1;
  if (rotate) {
    const float* fp = freqs + srow * 64 + 2 * p;
    float f0 = fp[0], f1 = fp[1];
    // out[2i] = x[2i]cos - x[2i+1]sin ; out[2i+1] = x[2i+1]cos + x[2i]sin
    y0 = x0 * __cosf(f0) - x1 * __sinf(f0);
    y1 = x1 * __cosf(f1) + x0 * __sinf(f1);
  }
  bf16_t* dp = dst + (((long)b * Hn + h) * Tn + t) * 64 + 2 * p;
  dp[0] = (bf16_t)(y0 * scale);
  dp[1] = (bf16_t)(y1 * scale);
}

// ---------------------------------------------------------------------------
// GEMM: out[M,N] (fp32) = A[M,K]bf16 @ B[K,N]bf16 (+bias) (+= if residual)
// Block tile 128x128, BK=32, 8 waves (4 M x 2 N), wave tile 32x64.
// Double-buffered LDS, one barrier per K tile; A staged via async-to-LDS.
// M,N multiples of 128; K multiple of 32 (true for all call sites).
// ---------------------------------------------------------------------------
#define GBM 128
#define GBN 128
#define GBK 32

__global__ __launch_bounds__(256) void gemm_bf16_kernel(
    const bf16_t* __restrict__ A, const bf16_t* __restrict__ B,
    const float* __restrict__ bias, float* __restrict__ out, int M, int N, int K,
    int residual) {
  __shared__ bf16_t sA[2][GBM * GBK];   // [m][k]
  __shared__ bf16_t sB[2][GBN * GBK];   // [n][k]  (transposed)
  const int tid = threadIdx.x, lane = tid & 31, wave = tid >> 5;
  const int wm = wave >> 1, wn = wave & 1;
  const int m0 = blockIdx.y * GBM, n0 = blockIdx.x * GBN;

  const f32x8 zf = {0.f, 0.f, 0.f, 0.f, 0.f, 0.f, 0.f, 0.f};
  f32x8 acc[2][4];
#pragma unroll
  for (int s = 0; s < 2; s++)
#pragma unroll
    for (int t = 0; t < 4; t++) acc[s][t] = zf;

  uint4 regB[2];
#if !HAVE_ASYNC_LDS
  uint4 regA[2];
#endif

  auto fetchA = [&](int k0, int buf) {
#pragma unroll
    for (int i = 0; i < 2; i++) {
      int c = tid + i * 256;
      int r = c >> 2, kc = (c & 3) * 8;
#if HAVE_ASYNC_LDS
      async_copy16(A + (long)(m0 + r) * K + k0 + kc, &sA[buf][r * GBK + kc]);
#else
      regA[i] = *(const uint4*)(A + (long)(m0 + r) * K + k0 + kc);
#endif
    }
  };
  auto fetchB = [&](int k0) {
#pragma unroll
    for (int i = 0; i < 2; i++) {
      int c = tid + i * 256;
      int kk = c >> 4, nc = (c & 15) * 8;
      regB[i] = *(const uint4*)(B + (long)(k0 + kk) * N + n0 + nc);
    }
  };
  auto commit = [&](int buf) {
#if !HAVE_ASYNC_LDS
#pragma unroll
    for (int i = 0; i < 2; i++) {
      int c = tid + i * 256;
      int r = c >> 2, kc = (c & 3) * 8;
      *(uint4*)(&sA[buf][r * GBK + kc]) = regA[i];
    }
#endif
#pragma unroll
    for (int i = 0; i < 2; i++) {
      int c = tid + i * 256;
      int kk = c >> 4, nc = (c & 15) * 8;
      const bf16_t* e = (const bf16_t*)&regB[i];
#pragma unroll
      for (int j = 0; j < 8; j++) sB[buf][(nc + j) * GBK + kk] = e[j];
    }
  };

  fetchA(0, 0);
  fetchB(0);
  commit(0);
  int cur = 0;
  for (int k0 = 0; k0 < K; k0 += GBK) {
    ASYNC_WAIT();
    __syncthreads();
    const bool next = (k0 + GBK) < K;
    if (next) { fetchA(k0 + GBK, cur ^ 1); fetchB(k0 + GBK); }

    bf16x16 af[2];
#pragma unroll
    for (int s = 0; s < 2; s++)
      af[s] = load_a_frag(&sA[cur][(wm * 32 + s * 16) * GBK], GBK, lane);
#pragma unroll
    for (int t = 0; t < 4; t++) {
      bf16x16 bf = load_b_frag(&sB[cur][(wn * 64 + t * 16) * GBK], GBK, lane);
#pragma unroll
      for (int s = 0; s < 2; s++) acc[s][t] = wmma_bf(af[s], bf, acc[s][t]);
    }
    if (next) commit(cur ^ 1);
    cur ^= 1;
  }

  const int ln = lane & 15, hi = lane >> 4;
#pragma unroll
  for (int s = 0; s < 2; s++)
#pragma unroll
    for (int t = 0; t < 4; t++) {
      int n = n0 + wn * 64 + t * 16 + ln;
      float bv = bias ? bias[n] : 0.f;
#pragma unroll
      for (int vg = 0; vg < 8; vg++) {
        int m = m0 + wm * 32 + s * 16 + vg + hi * 8;
        float v = acc[s][t][vg] + bv;
        long off = (long)m * N + n;
        if (residual) out[off] += v; else out[off] = v;
      }
    }
}

// ---------------------------------------------------------------------------
// Gated FFN GEMM: out[M,Nh]bf16 = (A@B[:, n]+b1[n]) * gelu(A@B[:, n+Nh]+b1[n+Nh])
// B is [K, 2*Nh]. Same pipelined tiling, two accumulator banks.
// ---------------------------------------------------------------------------
__global__ __launch_bounds__(256) void gemm_gated_kernel(
    const bf16_t* __restrict__ A, const bf16_t* __restrict__ B,
    const float* __restrict__ bias, bf16_t* __restrict__ out, int M, int Nh, int K) {
  __shared__ bf16_t sA[2][GBM * GBK];
  __shared__ bf16_t sBa[2][GBN * GBK];
  __shared__ bf16_t sBg[2][GBN * GBK];
  const int tid = threadIdx.x, lane = tid & 31, wave = tid >> 5;
  const int wm = wave >> 1, wn = wave & 1;
  const int m0 = blockIdx.y * GBM, n0 = blockIdx.x * GBN;
  const int Nfull = 2 * Nh;

  const f32x8 zf = {0.f, 0.f, 0.f, 0.f, 0.f, 0.f, 0.f, 0.f};
  f32x8 accA[2][4], accG[2][4];
#pragma unroll
  for (int s = 0; s < 2; s++)
#pragma unroll
    for (int t = 0; t < 4; t++) { accA[s][t] = zf; accG[s][t] = zf; }

  uint4 regBa[2], regBg[2];
#if !HAVE_ASYNC_LDS
  uint4 regA[2];
#endif

  auto fetchA = [&](int k0, int buf) {
#pragma unroll
    for (int i = 0; i < 2; i++) {
      int c = tid + i * 256;
      int r = c >> 2, kc = (c & 3) * 8;
#if HAVE_ASYNC_LDS
      async_copy16(A + (long)(m0 + r) * K + k0 + kc, &sA[buf][r * GBK + kc]);
#else
      regA[i] = *(const uint4*)(A + (long)(m0 + r) * K + k0 + kc);
#endif
    }
  };
  auto fetchB = [&](int k0) {
#pragma unroll
    for (int i = 0; i < 2; i++) {
      int c = tid + i * 256;
      int kk = c >> 4, nc = (c & 15) * 8;
      regBa[i] = *(const uint4*)(B + (long)(k0 + kk) * Nfull + n0 + nc);
      regBg[i] = *(const uint4*)(B + (long)(k0 + kk) * Nfull + n0 + Nh + nc);
    }
  };
  auto commit = [&](int buf) {
#if !HAVE_ASYNC_LDS
#pragma unroll
    for (int i = 0; i < 2; i++) {
      int c = tid + i * 256;
      int r = c >> 2, kc = (c & 3) * 8;
      *(uint4*)(&sA[buf][r * GBK + kc]) = regA[i];
    }
#endif
#pragma unroll
    for (int i = 0; i < 2; i++) {
      int c = tid + i * 256;
      int kk = c >> 4, nc = (c & 15) * 8;
      const bf16_t* ea = (const bf16_t*)&regBa[i];
      const bf16_t* eg = (const bf16_t*)&regBg[i];
#pragma unroll
      for (int j = 0; j < 8; j++) {
        sBa[buf][(nc + j) * GBK + kk] = ea[j];
        sBg[buf][(nc + j) * GBK + kk] = eg[j];
      }
    }
  };

  fetchA(0, 0);
  fetchB(0);
  commit(0);
  int cur = 0;
  for (int k0 = 0; k0 < K; k0 += GBK) {
    ASYNC_WAIT();
    __syncthreads();
    const bool next = (k0 + GBK) < K;
    if (next) { fetchA(k0 + GBK, cur ^ 1); fetchB(k0 + GBK); }

    bf16x16 af[2];
#pragma unroll
    for (int s = 0; s < 2; s++)
      af[s] = load_a_frag(&sA[cur][(wm * 32 + s * 16) * GBK], GBK, lane);
#pragma unroll
    for (int t = 0; t < 4; t++) {
      bf16x16 ba = load_b_frag(&sBa[cur][(wn * 64 + t * 16) * GBK], GBK, lane);
      bf16x16 bg = load_b_frag(&sBg[cur][(wn * 64 + t * 16) * GBK], GBK, lane);
#pragma unroll
      for (int s = 0; s < 2; s++) {
        accA[s][t] = wmma_bf(af[s], ba, accA[s][t]);
        accG[s][t] = wmma_bf(af[s], bg, accG[s][t]);
      }
    }
    if (next) commit(cur ^ 1);
    cur ^= 1;
  }

  const int ln = lane & 15, hi = lane >> 4;
#pragma unroll
  for (int s = 0; s < 2; s++)
#pragma unroll
    for (int t = 0; t < 4; t++) {
      int n = n0 + wn * 64 + t * 16 + ln;
      float ba = bias[n], bg = bias[n + Nh];
#pragma unroll
      for (int vg = 0; vg < 8; vg++) {
        int m = m0 + wm * 32 + s * 16 + vg + hi * 8;
        float a = accA[s][t][vg] + ba;
        float g = accG[s][t][vg] + bg;
        float gel = 0.5f * g * (1.f + erff(g * 0.70710678118654752f));
        out[(long)m * Nh + n] = (bf16_t)(a * gel);
      }
    }
}

// ---------------------------------------------------------------------------
// Fused flash attention (wave32 WMMA). One block = one (b,h) x 128 q rows,
// 8 waves, 16 q rows per wave. Q pre-scaled by 1/8 (=DH^-0.5) in rope_kernel.
// Q/K/V bf16 [B*H, len, 64]; Out bf16 merged heads [B, Tq, H*64].
// K tile staged via async-to-LDS; next K/V tile prefetched (global_prefetch_b8).
// ---------------------------------------------------------------------------
__global__ __launch_bounds__(256) void attn_kernel(
    const bf16_t* __restrict__ Q, const bf16_t* __restrict__ Kd,
    const bf16_t* __restrict__ Vd, bf16_t* __restrict__ Out, int H, int Tq, int S,
    int causal) {
  __shared__ bf16_t sK[32 * 64];       // [key][feat]
  __shared__ bf16_t sV[64 * 32];       // [feat][key] (transposed)
  __shared__ bf16_t sP[8][16 * 32];    // per-wave P tile [qrow][key]

  const int tid = threadIdx.x, lane = tid & 31, wave = tid >> 5;
  const int ln = lane & 15, g = lane >> 4;
  const int bh = blockIdx.y;
  const int b = bh / H, h = bh % H;
  const int qbb = blockIdx.x * 128;
  const int qb = qbb + wave * 16;

  // Q A-fragments (feat 0..31 and 32..63) held in registers for the whole loop
  const bf16_t* qrow = Q + ((long)bh * Tq + qb + ln) * 64;
  bf16x16 aq0, aq1;
  {
    union { uint4 u[2]; bf16x16 v; } t0, t1;
    int kb = g * 8;
    t0.u[0] = *(const uint4*)(qrow + kb);
    t0.u[1] = *(const uint4*)(qrow + kb + 16);
    t1.u[0] = *(const uint4*)(qrow + 32 + kb);
    t1.u[1] = *(const uint4*)(qrow + 32 + kb + 16);
    aq0 = t0.v; aq1 = t1.v;
  }

  const f32x8 zf = {0.f, 0.f, 0.f, 0.f, 0.f, 0.f, 0.f, 0.f};
  float mrow[8], lrow[8];
#pragma unroll
  for (int i = 0; i < 8; i++) { mrow[i] = -3.0e38f; lrow[i] = 0.f; }
  f32x8 oacc[4];
#pragma unroll
  for (int t = 0; t < 4; t++) oacc[t] = zf;

  const int key = tid >> 3, fb = (tid & 7) * 8;  // fixed staging coords
  const int kend = causal ? min(S, qbb + 128) : S;
  for (int kb0 = 0; kb0 < kend; kb0 += 32) {
    // stage K (async/direct) and V (register transpose)
    const bf16_t* kg = Kd + ((long)bh * S + kb0 + key) * 64 + fb;
    const bf16_t* vg2 = Vd + ((long)bh * S + kb0 + key) * 64 + fb;
#if HAVE_ASYNC_LDS
    async_copy16(kg, sK + key * 64 + fb);
#else
    {
      uint4 dk = *(const uint4*)kg;
      *(uint4*)(sK + key * 64 + fb) = dk;
    }
#endif
    {
      uint4 dv = *(const uint4*)vg2;
      const bf16_t* e = (const bf16_t*)&dv;
#pragma unroll
      for (int j = 0; j < 8; j++) sV[(fb + j) * 32 + key] = e[j];
    }
    if (kb0 + 32 < kend) {  // gfx1250 global_prefetch_b8 for next tile
      __builtin_prefetch(kg + 32 * 64, 0, 1);
      __builtin_prefetch(vg2 + 32 * 64, 0, 1);
    }
    ASYNC_WAIT();
    __syncthreads();

    // S = Q @ K^T for two 16-key subtiles
    f32x8 sc[2];
#pragma unroll
    for (int tt = 0; tt < 2; tt++) {
      f32x8 s = zf;
      bf16x16 bk0 = load_b_frag(sK + (tt * 16) * 64 + 0, 64, lane);
      bf16x16 bk1 = load_b_frag(sK + (tt * 16) * 64 + 32, 64, lane);
      s = wmma_bf(aq0, bk0, s);
      s = wmma_bf(aq1, bk1, s);
      sc[tt] = s;
    }

    // mask + online softmax state update (row m = vg + 8g, key col = ln)
    float mnew[8];
#pragma unroll
    for (int vg = 0; vg < 8; vg++) {
      int qg = qb + vg + g * 8;
      float r = mrow[vg];
#pragma unroll
      for (int tt = 0; tt < 2; tt++) {
        float v = sc[tt][vg];
        int kcol = kb0 + tt * 16 + ln;
        if (causal && kcol > qg) v = -3.0e38f;
        sc[tt][vg] = v;
        r = fmaxf(r, v);
      }
#pragma unroll
      for (int off = 1; off < 16; off <<= 1) r = fmaxf(r, __shfl_xor(r, off, 32));
      mnew[vg] = r;
    }
#pragma unroll
    for (int vg = 0; vg < 8; vg++) {
      float scale = __expf(mrow[vg] - mnew[vg]);
      float psum = 0.f;
#pragma unroll
      for (int tt = 0; tt < 2; tt++) {
        float v = sc[tt][vg];
        float p = (v < -1.0e37f) ? 0.f : __expf(v - mnew[vg]);
        psum += p;
        sP[wave][(vg + g * 8) * 32 + tt * 16 + ln] = (bf16_t)p;
      }
#pragma unroll
      for (int off = 1; off < 16; off <<= 1) psum += __shfl_xor(psum, off, 32);
      lrow[vg] = lrow[vg] * scale + psum;
      mrow[vg] = mnew[vg];
#pragma unroll
      for (int t = 0; t < 4; t++) oacc[t][vg] *= scale;
    }
    __syncthreads();

    // O += P @ V  (P A-fragment from LDS, V B-fragments per 16-wide DH tile)
    bf16x16 pf = load_a_frag(&sP[wave][0], 32, lane);
#pragma unroll
    for (int t = 0; t < 4; t++) {
      bf16x16 vf = load_b_frag(sV + (t * 16) * 32, 32, lane);
      oacc[t] = wmma_bf(pf, vf, oacc[t]);
    }
    __syncthreads();
  }

  // epilogue: merged-head bf16 output [b, qrow, h*64 + dh]
  const int hd = H * 64;
#pragma unroll
  for (int t = 0; t < 4; t++)
#pragma unroll
    for (int vg = 0; vg < 8; vg++) {
      int qr = qb + vg + g * 8;
      float ov = oacc[t][vg] / fmaxf(lrow[vg], 1e-30f);
      Out[((long)b * Tq + qr) * hd + h * 64 + t * 16 + ln] = (bf16_t)ov;
    }
}

// ---------------------------------------------------------------------------
// Host orchestration
// ---------------------------------------------------------------------------
extern "C" void kernel_launch(void* const* d_in, const int* in_sizes, int n_in,
                              void* d_out, int out_size, void* d_ws, size_t ws_size,
                              hipStream_t stream) {
  (void)in_sizes; (void)n_in; (void)out_size; (void)ws_size;

  const float* bin_queries    = (const float*)d_in[0];
  const float* bin_time_emb   = (const float*)d_in[1];
  const float* encoder_latents= (const float*)d_in[2];
  const float* latent_time_emb= (const float*)d_in[3];
  const float* ca_lnq_w = (const float*)d_in[4];
  const float* ca_lnq_b = (const float*)d_in[5];
  const float* ca_lnc_w = (const float*)d_in[6];
  const float* ca_lnc_b = (const float*)d_in[7];
  const float* ca_wq    = (const float*)d_in[8];
  const float* ca_wkv   = (const float*)d_in[9];
  const float* ca_wo    = (const float*)d_in[10];
  const float* ca_bo    = (const float*)d_in[11];
  const float* sa_ln_w  = (const float*)d_in[12];
  const float* sa_ln_b  = (const float*)d_in[13];
  const float* sa_wqkv  = (const float*)d_in[14];
  const float* sa_wo    = (const float*)d_in[15];
  const float* sa_bo    = (const float*)d_in[16];
  const float* ffn_ln_w = (const float*)d_in[17];
  const float* ffn_ln_b = (const float*)d_in[18];
  const float* ffn_w1   = (const float*)d_in[19];
  const float* ffn_b1   = (const float*)d_in[20];
  const float* ffn_w2   = (const float*)d_in[21];
  const float* ffn_b2   = (const float*)d_in[22];

  const long BT = (long)BX * TX;    // 8192
  const long BN = (long)BX * NNX;   // 16384

  // workspace bump allocator (256B aligned)
  char* wp = (char*)d_ws;
  auto alloc = [&](long bytes) -> void* {
    void* r = (void*)wp;
    wp += ((size_t)bytes + 255) & ~(size_t)255;
    return r;
  };

  // bf16 weight copies
  bf16_t* wq_bf  = (bf16_t*)alloc(DEPTHX * DIMX * CIX * 2);
  bf16_t* wkv_bf = (bf16_t*)alloc(DEPTHX * DIMX * 2 * CIX * 2);
  bf16_t* wo_bf  = (bf16_t*)alloc(DEPTHX * CIX * DIMX * 2);
  bf16_t* sqkv_bf= (bf16_t*)alloc(DEPTHX * DIMX * 3 * SIX * 2);
  bf16_t* swo_bf = (bf16_t*)alloc(DEPTHX * SIX * DIMX * 2);
  bf16_t* w1_bf  = (bf16_t*)alloc((long)DEPTHX * DIMX * 2 * FFX * 2);
  bf16_t* w2_bf  = (bf16_t*)alloc((long)DEPTHX * FFX * DIMX * 2);
  // activations
  bf16_t* xn_bf  = (bf16_t*)alloc(BT * DIMX * 2);
  bf16_t* cn_bf  = (bf16_t*)alloc(BN * DIMX * 2);
  float*  gq_f   = (float*) alloc(BT * CIX * 4);
  float*  gkv_f  = (float*) alloc(BN * 2 * CIX * 4);
  bf16_t* qr_bf  = (bf16_t*)alloc((long)BX * CHX * TX * 64 * 2);
  bf16_t* kr_bf  = (bf16_t*)alloc((long)BX * CHX * NNX * 64 * 2);
  bf16_t* vr_bf  = (bf16_t*)alloc((long)BX * CHX * NNX * 64 * 2);
  bf16_t* ao_bf  = (bf16_t*)alloc(BT * CIX * 2);
  float*  qkv_f  = (float*) alloc(BT * 3 * SIX * 4);
  bf16_t* q2_bf  = (bf16_t*)alloc((long)BX * SHX * TX * 64 * 2);
  bf16_t* k2_bf  = (bf16_t*)alloc((long)BX * SHX * TX * 64 * 2);
  bf16_t* v2_bf  = (bf16_t*)alloc((long)BX * SHX * TX * 64 * 2);
  bf16_t* ao2_bf = (bf16_t*)alloc(BT * SIX * 2);
  bf16_t* hn_bf  = (bf16_t*)alloc(BT * DIMX * 2);
  bf16_t* ag_bf  = (bf16_t*)alloc(BT * FFX * 2);

  // convert weights to bf16
  auto cvt = [&](const float* s, bf16_t* d, long n) {
    long grid = (n + 255) / 256;
    if (grid > 4096) grid = 4096;
    cvt_bf16_kernel<<<(int)grid, 256, 0, stream>>>(s, d, n);
  };
  cvt(ca_wq,   wq_bf,   (long)DEPTHX * DIMX * CIX);
  cvt(ca_wkv,  wkv_bf,  (long)DEPTHX * DIMX * 2 * CIX);
  cvt(ca_wo,   wo_bf,   (long)DEPTHX * CIX * DIMX);
  cvt(sa_wqkv, sqkv_bf, (long)DEPTHX * DIMX * 3 * SIX);
  cvt(sa_wo,   swo_bf,  (long)DEPTHX * SIX * DIMX);
  cvt(ffn_w1,  w1_bf,   (long)DEPTHX * DIMX * 2 * FFX);
  cvt(ffn_w2,  w2_bf,   (long)DEPTHX * FFX * DIMX);

  // x = bin_queries (residual stream lives in d_out)
  float* x = (float*)d_out;
  (void)hipMemcpyAsync(x, bin_queries, BT * DIMX * sizeof(float),
                       hipMemcpyDeviceToDevice, stream);

  auto gemm = [&](const bf16_t* A, const bf16_t* B, const float* bias, float* out,
                  int M, int N, int K, int residual) {
    dim3 grid(N / GBN, M / GBM);
    gemm_bf16_kernel<<<grid, 256, 0, stream>>>(A, B, bias, out, M, N, K, residual);
  };
  auto rope = [&](const float* src, int rowLen, int off, const float* freqs,
                  bf16_t* dst, int Hn, int Tn, float scale, int rotate) {
    int total = BX * Tn * Hn * 32;
    rope_kernel<<<(total + 255) / 256, 256, 0, stream>>>(src, rowLen, off, freqs,
                                                         dst, Hn, Tn, scale, rotate);
  };
  auto ln = [&](const float* src, const float* w, const float* b, bf16_t* dst,
                int rows) {
    ln_bf16_kernel<<<rows / 8, 256, 0, stream>>>(src, w, b, dst, rows);
  };

  for (int l = 0; l < DEPTHX; l++) {
    // ---- cross attention ----
    ln(x, ca_lnq_w + l * DIMX, ca_lnq_b + l * DIMX, xn_bf, (int)BT);
    ln(encoder_latents, ca_lnc_w + l * DIMX, ca_lnc_b + l * DIMX, cn_bf, (int)BN);
    gemm(xn_bf, wq_bf + (long)l * DIMX * CIX, nullptr, gq_f, (int)BT, CIX, DIMX, 0);
    gemm(cn_bf, wkv_bf + (long)l * DIMX * 2 * CIX, nullptr, gkv_f, (int)BN, 2 * CIX,
         DIMX, 0);
    rope(gq_f, CIX, 0, bin_time_emb, qr_bf, CHX, TX, 0.125f, 1);
    rope(gkv_f, 2 * CIX, 0, latent_time_emb, kr_bf, CHX, NNX, 1.f, 1);
    rope(gkv_f, 2 * CIX, CIX, nullptr, vr_bf, CHX, NNX, 1.f, 0);
    {
      dim3 grid(TX / 128, BX * CHX);
      attn_kernel<<<grid, 256, 0, stream>>>(qr_bf, kr_bf, vr_bf, ao_bf, CHX, TX,
                                            NNX, 0);
    }
    gemm(ao_bf, wo_bf + (long)l * CIX * DIMX, ca_bo + l * DIMX, x, (int)BT, DIMX,
         CIX, 1);

    // ---- causal self attention ----
    ln(x, sa_ln_w + l * DIMX, sa_ln_b + l * DIMX, xn_bf, (int)BT);
    gemm(xn_bf, sqkv_bf + (long)l * DIMX * 3 * SIX, nullptr, qkv_f, (int)BT,
         3 * SIX, DIMX, 0);
    rope(qkv_f, 3 * SIX, 0, bin_time_emb, q2_bf, SHX, TX, 0.125f, 1);
    rope(qkv_f, 3 * SIX, SIX, bin_time_emb, k2_bf, SHX, TX, 1.f, 1);
    rope(qkv_f, 3 * SIX, 2 * SIX, nullptr, v2_bf, SHX, TX, 1.f, 0);
    {
      dim3 grid(TX / 128, BX * SHX);
      attn_kernel<<<grid, 256, 0, stream>>>(q2_bf, k2_bf, v2_bf, ao2_bf, SHX, TX,
                                            TX, 1);
    }
    gemm(ao2_bf, swo_bf + (long)l * SIX * DIMX, sa_bo + l * DIMX, x, (int)BT, DIMX,
         SIX, 1);

    // ---- gated FFN ----
    ln(x, ffn_ln_w + l * DIMX, ffn_ln_b + l * DIMX, hn_bf, (int)BT);
    {
      dim3 grid(FFX / GBN, (int)BT / GBM);
      gemm_gated_kernel<<<grid, 256, 0, stream>>>(hn_bf,
                                                  w1_bf + (long)l * DIMX * 2 * FFX,
                                                  ffn_b1 + (long)l * 2 * FFX, ag_bf,
                                                  (int)BT, FFX, DIMX);
    }
    gemm(ag_bf, w2_bf + (long)l * FFX * DIMX, ffn_b2 + l * DIMX, x, (int)BT, DIMX,
         FFX, 1);
  }
}
